// PointCloudDiffusionModel_72490458022544
// MI455X (gfx1250) — matrix operand order repaired and checked by reading, not compiled, not run on hardware
//
#include <hip/hip_runtime.h>
#include <hip/hip_bf16.h>

typedef __attribute__((ext_vector_type(16))) _Float16 v16h;
typedef __attribute__((ext_vector_type(8)))  _Float16 v8h;
typedef __attribute__((ext_vector_type(8)))  float    v8f;

#define T_STEPS 1000
#define PI_D 3.14159265358979323846

// ---------------------------------------------------------------------------
// Kernel 0: cosine beta schedule (reference computes this host-side in fp64)
// ---------------------------------------------------------------------------
__global__ void schedule_kernel(float* __restrict__ sched) {
    if (threadIdx.x == 0 && blockIdx.x == 0) {
        const double s = 0.008;
        double c0 = cos((0.0 / T_STEPS + s) / (1.0 + s) * PI_D * 0.5);
        double f0 = c0 * c0;
        double acprev = 1.0;
        double prod = 1.0;
        for (int i = 1; i <= T_STEPS; ++i) {
            double ci = cos(((double)i / T_STEPS + s) / (1.0 + s) * PI_D * 0.5);
            double ac = (ci * ci) / f0;
            double beta = 1.0 - ac / acprev;
            if (beta < 1e-4) beta = 1e-4;
            if (beta > 2e-2) beta = 2e-2;
            prod *= (1.0 - beta);
            acprev = ac;
            sched[i - 1]           = (float)sqrt(prod);
            sched[T_STEPS + i - 1] = (float)sqrt(1.0 - prod);
        }
    }
}

// ---------------------------------------------------------------------------
// Convert a f32 weight matrix to f16 in workspace (done once per call).
// ---------------------------------------------------------------------------
__global__ __launch_bounds__(256)
void cvt_w_kernel(const float* __restrict__ src, _Float16* __restrict__ dst, int n) {
    int i = blockIdx.x * 256 + threadIdx.x;
    if (i < n) dst[i] = (_Float16)src[i];
}

// ---------------------------------------------------------------------------
// Kernel 1: q_sample + concat t_emb + layer1 (4 -> 64, VALU; too thin for WMMA)
// ---------------------------------------------------------------------------
__global__ __launch_bounds__(256)
void stage0_kernel(const float* __restrict__ x, const int* __restrict__ t,
                   const float* __restrict__ noise,
                   const float* __restrict__ w1, const float* __restrict__ b1,
                   const float* __restrict__ sched,
                   _Float16* __restrict__ Y1, int P, int Npts) {
    __shared__ float wl[64 * 4];
    __shared__ float bl[64];
    for (int i = threadIdx.x; i < 256; i += 256) wl[i] = w1[i];
    if (threadIdx.x < 64) bl[threadIdx.x] = b1[threadIdx.x];
    __syncthreads();

    int p = blockIdx.x * 256 + threadIdx.x;
    if (p >= P) return;
    int b  = p / Npts;
    int ti = t[b];
    float sa = sched[ti];
    float s1 = sched[T_STEPS + ti];
    float h0 = sa * x[p * 3 + 0] + s1 * noise[p * 3 + 0];
    float h1 = sa * x[p * 3 + 1] + s1 * noise[p * 3 + 1];
    float h2 = sa * x[p * 3 + 2] + s1 * noise[p * 3 + 2];
    float h3 = (float)ti;
    _Float16* out = Y1 + (size_t)p * 64;
    #pragma unroll
    for (int o8 = 0; o8 < 64; o8 += 8) {
        v8h outv;
        #pragma unroll
        for (int j = 0; j < 8; ++j) {
            int o = o8 + j;
            float acc = bl[o] + wl[o * 4 + 0] * h0 + wl[o * 4 + 1] * h1
                              + wl[o * 4 + 2] * h2 + wl[o * 4 + 3] * h3;
            outv[j] = (_Float16)acc;
        }
        *(v8h*)(out + o8) = outv;
    }
}

// ---------------------------------------------------------------------------
// Per-channel sum / sum^2 partials (deterministic: no atomics, fixed order).
// ---------------------------------------------------------------------------
template <int Co>
__global__ __launch_bounds__(256)
void stats_kernel(const _Float16* __restrict__ Y,
                  float* __restrict__ psum, float* __restrict__ psq,
                  int P, int NB) {
    constexpr int CG = Co / 8;      // column groups of 8 channels
    constexpr int G  = 256 / CG;    // point-stride groups
    __shared__ float lsum[256 * 8];
    __shared__ float lsq[256 * 8];
    const int cg = threadIdx.x % CG;
    const int g  = threadIdx.x / CG;
    const int ppb = P / NB;
    const int p0 = blockIdx.x * ppb;
    float s[8], q[8];
    #pragma unroll
    for (int e = 0; e < 8; ++e) { s[e] = 0.f; q[e] = 0.f; }
    for (int p = p0 + g; p < p0 + ppb; p += G) {
        v8h yv = *(const v8h*)(Y + (size_t)p * Co + cg * 8);
        #pragma unroll
        for (int e = 0; e < 8; ++e) {
            float v = (float)yv[e];
            s[e] += v;
            q[e] += v * v;
        }
    }
    #pragma unroll
    for (int e = 0; e < 8; ++e) {
        lsum[threadIdx.x * 8 + e] = s[e];
        lsq[threadIdx.x * 8 + e]  = q[e];
    }
    __syncthreads();
    if (threadIdx.x < Co) {
        const int c = threadIdx.x;
        const int ccg = c / 8, e = c % 8;
        float ts = 0.f, tq = 0.f;
        #pragma unroll 4
        for (int gg = 0; gg < G; ++gg) {
            int tx = gg * CG + ccg;
            ts += lsum[tx * 8 + e];
            tq += lsq[tx * 8 + e];
        }
        psum[blockIdx.x * Co + c] = ts;
        psq[blockIdx.x * Co + c]  = tq;
    }
}

// ---------------------------------------------------------------------------
// Finalize BN: scale = gamma / sqrt(var + eps), shift = beta - mean*scale.
// ---------------------------------------------------------------------------
__global__ void bn_finalize(const float* __restrict__ psum, const float* __restrict__ psq,
                            const float* __restrict__ gamma, const float* __restrict__ beta,
                            float* __restrict__ scale, float* __restrict__ shift,
                            int Co, int NB, float invP) {
    int c = threadIdx.x;
    if (c < Co) {
        float s = 0.f, q = 0.f;
        for (int b = 0; b < NB; ++b) {
            s += psum[b * Co + c];
            q += psq[b * Co + c];
        }
        float m = s * invP;
        float v = q * invP - m * m;
        float sc = gamma[c] * __frsqrt_rn(v + 1e-5f);
        scale[c] = sc;
        shift[c] = beta[c] - m * sc;
    }
}

// ---------------------------------------------------------------------------
// WMMA GEMM: Yout(P x Co) = relu(bn(Yin)) * W^T + bias
// A = weight tile (M=16 out-ch x K=32 in-ch), B = activation tile (K x N=16 pts).
// BN affine + ReLU fully in packed f16 (v_pk_fma_f16 / v_max_f16).
// Two point-tiles per weight fetch: halves weight-load rate, and the two
// independent accumulator chains keep the XDL pipe busy without hazard NOPs.
// ---------------------------------------------------------------------------
template <int Ci, int Co>
__global__ __launch_bounds__(256)
void gemm_bn_relu_wmma(const _Float16* __restrict__ Yin,   // P x Ci (f16)
                       const _Float16* __restrict__ Wh,    // Co x Ci (f16)
                       const float* __restrict__ bias,     // Co
                       const float* __restrict__ scale,    // Ci (BN of input)
                       const float* __restrict__ shift,    // Ci
                       _Float16* __restrict__ Yout,        // P x Co
                       int P) {
    constexpr int KC = Ci / 32;
    constexpr int MT = Co / 16;
    __shared__ __align__(32) _Float16 sch[Ci];
    __shared__ __align__(32) _Float16 shh[Ci];
    __shared__ __align__(32) float bl[Co];
    for (int i = threadIdx.x; i < Ci; i += 256) {
        sch[i] = (_Float16)scale[i];
        shh[i] = (_Float16)shift[i];
    }
    for (int i = threadIdx.x; i < Co; i += 256) bl[i] = bias[i];
    __syncthreads();

    const int lane = threadIdx.x & 31;
    const int wave = threadIdx.x >> 5;
    const int hi   = lane >> 4;
    const int l16  = lane & 15;
    const int nPairs = P >> 5;                 // pairs of 16-point tiles
    const int stride = gridDim.x * 8;

    for (int pr = blockIdx.x * 8 + wave; pr < nPairs; pr += stride) {
        const int prow0 = (pr << 5) + l16;             // tile 0 point row
        const int prow1 = prow0 + 16;                  // tile 1 point row
        const _Float16* arow0 = Yin + (size_t)prow0 * Ci;
        const _Float16* arow1 = Yin + (size_t)prow1 * Ci;

        // Prefetch next pair's activation rows (global_prefetch_b8 path).
        if (pr + stride < nPairs) {
            const _Float16* nx = Yin + ((size_t)((pr + stride) << 5) + l16) * Ci;
            __builtin_prefetch(nx, 0, 1);
        }

        // Load B (activation) fragments; BN + ReLU in packed f16.
        v16h bact0[KC], bact1[KC];
        #pragma unroll
        for (int ko = 0; ko < KC; ++ko) {
            const int k0 = ko * 32 + hi * 16;
            v16h scv = *(const v16h*)&sch[k0];
            v16h shv = *(const v16h*)&shh[k0];
            v16h r0 = (*(const v16h*)(arow0 + k0)) * scv + shv;
            v16h r1 = (*(const v16h*)(arow1 + k0)) * scv + shv;
            #pragma unroll
            for (int j = 0; j < 16; ++j) {
                r0[j] = __builtin_fmaxf16(r0[j], (_Float16)0.f);
                r1[j] = __builtin_fmaxf16(r1[j], (_Float16)0.f);
            }
            bact0[ko] = r0;
            bact1[ko] = r1;
        }

        _Float16* outp0 = Yout + (size_t)prow0 * Co;
        _Float16* outp1 = Yout + (size_t)prow1 * Co;
        #pragma unroll
        for (int mt = 0; mt < MT; ++mt) {
            // Weight (A) fragments for this out-channel tile, all K chunks.
            v16h wf[KC];
            #pragma unroll
            for (int ko = 0; ko < KC; ++ko) {
                const _Float16* wrow = Wh + (size_t)(mt * 16 + l16) * Ci + ko * 32 + hi * 16;
                v8h w0 = *(const v8h*)(wrow);
                v8h w1 = *(const v8h*)(wrow + 8);
                #pragma unroll
                for (int j = 0; j < 8; ++j) { wf[ko][j] = w0[j]; wf[ko][8 + j] = w1[j]; }
            }

            v8f bv = *(const v8f*)&bl[mt * 16 + 8 * hi];
            v8f acc0 = bv;
            v8f acc1 = bv;
            #pragma unroll
            for (int ko = 0; ko < KC; ++ko) {
                acc0 = __builtin_amdgcn_wmma_f32_16x16x32_f16(
                        false, wf[ko], false, bact0[ko], (short)0, acc0, false, false);
                acc1 = __builtin_amdgcn_wmma_f32_16x16x32_f16(
                        false, wf[ko], false, bact1[ko], (short)0, acc1, false, false);
            }

            v8h o0, o1;
            #pragma unroll
            for (int r = 0; r < 8; ++r) { o0[r] = (_Float16)acc0[r]; o1[r] = (_Float16)acc1[r]; }
            *(v8h*)(outp0 + mt * 16 + 8 * hi) = o0;
            *(v8h*)(outp1 + mt * 16 + 8 * hi) = o1;
        }
    }
}

// ---------------------------------------------------------------------------
// Final: bn5 + relu, layer6 (64 -> 3, VALU), squared error vs noise,
// deterministic block partial sums.
// ---------------------------------------------------------------------------
__global__ __launch_bounds__(256)
void final_kernel(const _Float16* __restrict__ Y5,
                  const float* __restrict__ scale, const float* __restrict__ shift,
                  const float* __restrict__ w6, const float* __restrict__ b6,
                  const float* __restrict__ noise,
                  float* __restrict__ lpart, int P) {
    __shared__ float wl[3 * 64];
    __shared__ float bl[3];
    __shared__ float sc[64];
    __shared__ float sh[64];
    for (int i = threadIdx.x; i < 192; i += 256) wl[i] = w6[i];
    if (threadIdx.x < 3) bl[threadIdx.x] = b6[threadIdx.x];
    if (threadIdx.x < 64) { sc[threadIdx.x] = scale[threadIdx.x]; sh[threadIdx.x] = shift[threadIdx.x]; }
    __syncthreads();

    float acc = 0.f;
    for (int p = blockIdx.x * 256 + threadIdx.x; p < P; p += gridDim.x * 256) {
        const _Float16* yr = Y5 + (size_t)p * 64;
        float av[64];
        #pragma unroll
        for (int c8 = 0; c8 < 64; c8 += 8) {
            v8h yv = *(const v8h*)(yr + c8);
            #pragma unroll
            for (int j = 0; j < 8; ++j) {
                float v = (float)yv[j] * sc[c8 + j] + sh[c8 + j];
                av[c8 + j] = v > 0.f ? v : 0.f;
            }
        }
        #pragma unroll
        for (int o = 0; o < 3; ++o) {
            float d = bl[o];
            #pragma unroll
            for (int c = 0; c < 64; ++c) d += wl[o * 64 + c] * av[c];
            d -= noise[p * 3 + o];
            acc += d * d;
        }
    }
    __shared__ float red[256];
    red[threadIdx.x] = acc;
    __syncthreads();
    for (int s2 = 128; s2 > 0; s2 >>= 1) {
        if (threadIdx.x < s2) red[threadIdx.x] += red[threadIdx.x + s2];
        __syncthreads();
    }
    if (threadIdx.x == 0) lpart[blockIdx.x] = red[0];
}

__global__ void loss_finalize(const float* __restrict__ lpart, int nb,
                              float invTotal, float* __restrict__ out) {
    if (threadIdx.x == 0 && blockIdx.x == 0) {
        float s = 0.f;
        for (int i = 0; i < nb; ++i) s += lpart[i];
        out[0] = s * invTotal;
    }
}

// ---------------------------------------------------------------------------
// Host launcher
// ---------------------------------------------------------------------------
extern "C" void kernel_launch(void* const* d_in, const int* in_sizes, int n_in,
                              void* d_out, int out_size, void* d_ws, size_t ws_size,
                              hipStream_t stream) {
    const float* x     = (const float*)d_in[0];
    const int*   t     = (const int*)  d_in[1];
    const float* noise = (const float*)d_in[2];
    const float* w1 = (const float*)d_in[3];  const float* b1 = (const float*)d_in[4];
    const float* w2 = (const float*)d_in[5];  const float* b2 = (const float*)d_in[6];
    const float* w3 = (const float*)d_in[7];  const float* b3 = (const float*)d_in[8];
    const float* w4 = (const float*)d_in[9];  const float* b4 = (const float*)d_in[10];
    const float* w5 = (const float*)d_in[11]; const float* b5 = (const float*)d_in[12];
    const float* w6 = (const float*)d_in[13]; const float* b6 = (const float*)d_in[14];
    const float* g1 = (const float*)d_in[15]; const float* be1 = (const float*)d_in[16];
    const float* g2 = (const float*)d_in[17]; const float* be2 = (const float*)d_in[18];
    const float* g3 = (const float*)d_in[19]; const float* be3 = (const float*)d_in[20];
    const float* g4 = (const float*)d_in[21]; const float* be4 = (const float*)d_in[22];
    const float* g5 = (const float*)d_in[23]; const float* be5 = (const float*)d_in[24];

    const int B    = in_sizes[1];
    const int P    = in_sizes[0] / 3;   // 524288
    const int Npts = P / B;             // 8192
    const int NB   = 1024;

    // Workspace layout (bytes)
    const size_t BUFSZ = (size_t)P * 128 * sizeof(_Float16); // 134 MB
    char* ws = (char*)d_ws;
    _Float16* bufA = (_Float16*)(ws);                        // Y1 / Y3 / Y5
    _Float16* bufB = (_Float16*)(ws + BUFSZ);                // Y2 / Y4
    float* sched   = (float*)(ws + 2 * BUFSZ);               // 8 KB
    float* psum    = (float*)(ws + 2 * BUFSZ + 8192);        // NB*128 f32
    float* psq     = psum + (size_t)NB * 128;
    float* scl     = psq  + (size_t)NB * 128;                // 128 f32
    float* shf     = scl + 128;                              // 128 f32
    float* lpart   = shf + 128;                              // NB f32
    _Float16* w2h  = (_Float16*)(lpart + NB);                // 128x64
    _Float16* w3h  = w2h + 128 * 64;                         // 128x128
    _Float16* w4h  = w3h + 128 * 128;                        // 128x128
    _Float16* w5h  = w4h + 128 * 128;                        // 64x128
    float* outp    = (float*)d_out;

    const float invP = 1.0f / (float)P;

    schedule_kernel<<<1, 32, 0, stream>>>(sched);
    cvt_w_kernel<<<(128 * 64  + 255) / 256, 256, 0, stream>>>(w2, w2h, 128 * 64);
    cvt_w_kernel<<<(128 * 128 + 255) / 256, 256, 0, stream>>>(w3, w3h, 128 * 128);
    cvt_w_kernel<<<(128 * 128 + 255) / 256, 256, 0, stream>>>(w4, w4h, 128 * 128);
    cvt_w_kernel<<<(64 * 128  + 255) / 256, 256, 0, stream>>>(w5, w5h, 64 * 128);

    // Layer 1: q_sample + concat + lin(4->64)
    stage0_kernel<<<(P + 255) / 256, 256, 0, stream>>>(x, t, noise, w1, b1, sched,
                                                       bufA, P, Npts);
    stats_kernel<64><<<NB, 256, 0, stream>>>(bufA, psum, psq, P, NB);
    bn_finalize<<<1, 128, 0, stream>>>(psum, psq, g1, be1, scl, shf, 64, NB, invP);

    // Layer 2: bn1+relu -> lin(64->128)          [WMMA]
    gemm_bn_relu_wmma<64, 128><<<1024, 256, 0, stream>>>(bufA, w2h, b2, scl, shf, bufB, P);
    stats_kernel<128><<<NB, 256, 0, stream>>>(bufB, psum, psq, P, NB);
    bn_finalize<<<1, 128, 0, stream>>>(psum, psq, g2, be2, scl, shf, 128, NB, invP);

    // Layer 3: bn2+relu -> lin(128->128)         [WMMA]
    gemm_bn_relu_wmma<128, 128><<<1024, 256, 0, stream>>>(bufB, w3h, b3, scl, shf, bufA, P);
    stats_kernel<128><<<NB, 256, 0, stream>>>(bufA, psum, psq, P, NB);
    bn_finalize<<<1, 128, 0, stream>>>(psum, psq, g3, be3, scl, shf, 128, NB, invP);

    // Layer 4: bn3+relu -> lin(128->128)         [WMMA]
    gemm_bn_relu_wmma<128, 128><<<1024, 256, 0, stream>>>(bufA, w4h, b4, scl, shf, bufB, P);
    stats_kernel<128><<<NB, 256, 0, stream>>>(bufB, psum, psq, P, NB);
    bn_finalize<<<1, 128, 0, stream>>>(psum, psq, g4, be4, scl, shf, 128, NB, invP);

    // Layer 5: bn4+relu -> lin(128->64)          [WMMA]
    gemm_bn_relu_wmma<128, 64><<<1024, 256, 0, stream>>>(bufB, w5h, b5, scl, shf, bufA, P);
    stats_kernel<64><<<NB, 256, 0, stream>>>(bufA, psum, psq, P, NB);
    bn_finalize<<<1, 128, 0, stream>>>(psum, psq, g5, be5, scl, shf, 64, NB, invP);

    // Layer 6 + MSE loss
    final_kernel<<<NB, 256, 0, stream>>>(bufA, scl, shf, w6, b6, noise, lpart, P);
    loss_finalize<<<1, 32, 0, stream>>>(lpart, NB, 1.0f / ((float)P * 3.0f), outp);
}
